// LSTM_MLP_20615843020954
// MI455X (gfx1250) — matrix operand order repaired and checked by reading, not compiled, not run on hardware
//
#include <hip/hip_runtime.h>

// ---------------------------------------------------------------------------
// LSTM(4 layers, H=64) + MLP head for MI455X (gfx1250, wave32, WMMA).
// Recurrent GEMM via v_wmma_f32_16x16x32_bf16; weights resident in VGPRs;
// x_t prefetched one step ahead via async global->LDS (double-buffered);
// h written back coalesced from LDS one step deferred.
// ---------------------------------------------------------------------------

typedef unsigned short u16;
typedef __attribute__((ext_vector_type(16))) __bf16   v16bf;
typedef __attribute__((ext_vector_type(8)))  float    v8f;
typedef __attribute__((ext_vector_type(8)))  unsigned v8u;

union FragU {
  unsigned a[8];
  v8u      u;
  v16bf    b;
};

#define HID     64
#define GATES   256
#define B_TOT   2048
#define T_TOT   512
#define LT_STRIDE 72   // LDS row stride in bf16 elems (avoids bank conflicts)

// ---- feature probes --------------------------------------------------------
#if __has_builtin(__builtin_amdgcn_tanhf)
#  define TANH_NATIVE(x) __builtin_amdgcn_tanhf(x)
#elif __has_builtin(__builtin_amdgcn_tanh_f32)
#  define TANH_NATIVE(x) __builtin_amdgcn_tanh_f32(x)
#endif

#if __has_builtin(__builtin_amdgcn_global_load_async_to_lds_b64)
#  define HAVE_ASYNC_LDS 1
// Builtin signature: (V2i addrspace(1)*, V2i addrspace(3)*, Ii offset, Ii cpol)
typedef int v2i_t __attribute__((vector_size(8)));
typedef __attribute__((address_space(1))) v2i_t* gld64_p;
typedef __attribute__((address_space(3))) v2i_t* lds64_p;
#endif

__device__ __forceinline__ u16 f2bf(float f) {
  unsigned u = __float_as_uint(f);
  return (u16)((u + 0x7FFFu + ((u >> 16) & 1u)) >> 16);   // RNE
}
__device__ __forceinline__ float bf2f(u16 h) {
  return __uint_as_float(((unsigned)h) << 16);
}
__device__ __forceinline__ float tanh_f(float x) {
#ifdef TANH_NATIVE
  return TANH_NATIVE(x);
#else
  return 2.f / (1.f + __expf(-2.f * x)) - 1.f;
#endif
}
__device__ __forceinline__ float sigm(float x) {
#ifdef TANH_NATIVE
  return fmaf(0.5f, TANH_NATIVE(0.5f * x), 0.5f);
#else
  return 1.f / (1.f + __expf(-x));
#endif
}

// A-matrix (16x32 bf16) fragment from an LDS tile [16][LT_STRIDE].
// Layout: lanes 0-15 -> m, half=lane/16; VGPR j: k = 16*half + 2j (contiguous, matches B).
__device__ __forceinline__ v16bf load_a(const u16* lds, int m, int half, int kf) {
  const u16* p = lds + m * LT_STRIDE + kf * 32 + half * 16;
  FragU f;
  *(uint2*)&f.a[0] = *(const uint2*)(p);        // k+0..3
  *(uint2*)&f.a[2] = *(const uint2*)(p + 4);    // k+4..7
  *(uint2*)&f.a[4] = *(const uint2*)(p + 8);    // k+8..11
  *(uint2*)&f.a[6] = *(const uint2*)(p + 12);   // k+12..15
  return f.b;
}

// ---------------------------------------------------------------------------
// Pre/post kernels
// ---------------------------------------------------------------------------

// fp32 x[B][T][18] -> bf16 seq[B][T][64] zero-padded
__global__ void cvt_input_k(const float* __restrict__ x, u16* __restrict__ seq) {
  long idx = (long)blockIdx.x * 256 + threadIdx.x;      // over B*T*64
  int  j   = (int)(idx & 63);
  long bt  = idx >> 6;
  float v  = (j < 18) ? x[bt * 18 + j] : 0.f;
  seq[idx] = f2bf(v);
}

// Pack fp32 weights -> bf16 [256][64] (Wi zero-padded to K=64), bias = bi+bh.
__global__ void pack_k(const float* __restrict__ Wi, const float* __restrict__ Wh,
                       const float* __restrict__ bi, const float* __restrict__ bh,
                       u16* __restrict__ wip, u16* __restrict__ whp,
                       float* __restrict__ bp, int din) {
  int idx = blockIdx.x * 256 + threadIdx.x;             // 0..16383
  int g = idx >> 6, k = idx & 63;
  wip[idx] = f2bf((k < din) ? Wi[g * din + k] : 0.f);
  whp[idx] = f2bf(Wh[idx]);
  if (k == 0) bp[g] = bi[g] + bh[g];
}

// MLP head: out = relu(h_last @ W1^T + b1) @ W2^T + b2
__global__ __launch_bounds__(128) void head_k(const u16* __restrict__ seq,
    const float* __restrict__ W1, const float* __restrict__ b1,
    const float* __restrict__ W2, const float* __restrict__ b2,
    float* __restrict__ out) {
  __shared__ float hv[HID];
  __shared__ float hid[128];
  int b = blockIdx.x, j = threadIdx.x;
  if (j < HID) hv[j] = bf2f(seq[((long)b * T_TOT + (T_TOT - 1)) * HID + j]);
  __syncthreads();
  float s = b1[j];
  for (int k = 0; k < HID; ++k) s += hv[k] * W1[j * HID + k];
  hid[j] = fmaxf(s, 0.f);
  __syncthreads();
  if (j < 3) {
    float s2 = b2[j];
    for (int k = 0; k < 128; ++k) s2 += hid[k] * W2[j * 128 + k];
    out[b * 3 + j] = s2;
  }
}

// ---------------------------------------------------------------------------
// LSTM layer, in-place on seq[B][T][64] (bf16).
// Grid: B/16 workgroups of 128 threads (4 waves). WG owns 16 batch rows.
// Wave q computes gate columns q*16..q*16+15 of i,f,g,o (N tiles q, q+4, q+8, q+12).
// x_{t+1} prefetched into double-buffered LDS via async-to-LDS during step t;
// h_t global write-back deferred one step, done coalesced from LDS.
// ---------------------------------------------------------------------------
__global__ __launch_bounds__(128) void lstm_layer_k(
    u16* __restrict__ seq,
    const u16* __restrict__ Wi,      // bf16 [256][64] row-major (g, k)
    const u16* __restrict__ Wh,      // bf16 [256][64]
    const float* __restrict__ bias)  // [256] = bi+bh
{
  __shared__ __align__(16) u16 ldsX[2][16 * LT_STRIDE];
  __shared__ __align__(16) u16 ldsH[16 * LT_STRIDE];

  const int tid  = threadIdx.x;
  const int lane = tid & 31;
  const int wv   = tid >> 5;        // wave id 0..3
  const int half = lane >> 4;
  const int nl   = lane & 15;
  const long b0  = (long)blockIdx.x * 16;

  // Stage x_t (16 rows x 64 bf16) into an LDS buffer (async when available).
  auto stage_x = [&](int t, u16* dstbuf) {
#pragma unroll
    for (int i2 = 0; i2 < 2; ++i2) {
      int i = tid + i2 * 128;
      int r = i >> 4, c = i & 15;
      u16* src = seq + ((b0 + r) * (long)T_TOT + t) * HID + c * 4;
      u16* dst = dstbuf + r * LT_STRIDE + c * 4;
#ifdef HAVE_ASYNC_LDS
      __builtin_amdgcn_global_load_async_to_lds_b64(
          (gld64_p)(v2i_t*)src, (lds64_p)(v2i_t*)dst, 0, 0);
#else
      *(uint2*)dst = *(const uint2*)src;
#endif
    }
  };

  // Preload B-matrix (weight) fragments: resident in VGPRs for all 512 steps.
  // B(32x16) layout: lane = n; VGPR j: k = 16*half + 2j  -> 32 contiguous bytes.
  FragU wif[4][2], whf[4][2];
#pragma unroll
  for (int nt = 0; nt < 4; ++nt) {
    const int g = nt * 64 + wv * 16 + nl;
#pragma unroll
    for (int kf = 0; kf < 2; ++kf) {
      wif[nt][kf].u = *(const v8u*)(Wi + g * HID + kf * 32 + half * 16);
      whf[nt][kf].u = *(const v8u*)(Wh + g * HID + kf * 32 + half * 16);
    }
  }

  // Bias pre-splat (loop-invariant C operand; WMMA allows D != C)
  v8f bias_acc[4];
#pragma unroll
  for (int nt = 0; nt < 4; ++nt) {
    float bv = bias[nt * 64 + wv * 16 + nl];
#pragma unroll
    for (int r = 0; r < 8; ++r) bias_acc[nt][r] = bv;
  }

  float cst[8];                      // cell state: 8 (m, j) elems per lane
#pragma unroll
  for (int r = 0; r < 8; ++r) cst[r] = 0.f;

  for (int i = tid; i < 16 * LT_STRIDE; i += 128) ldsH[i] = 0;

  // Prologue: stage x_0 into buffer 0.
  stage_x(0, ldsX[0]);

  for (int t = 0; t < T_TOT; ++t) {
    const u16* xbuf = ldsX[t & 1];
#ifdef HAVE_ASYNC_LDS
#  if __has_builtin(__builtin_amdgcn_s_wait_asynccnt)
    __builtin_amdgcn_s_wait_asynccnt(0);
#  else
    asm volatile("s_wait_asynccnt 0" ::: "memory");
#  endif
#endif
    __syncthreads();                 // B: x_t staged; h_{t-1} LDS writes visible

    // ---- prefetch x_{t+1} into the other buffer (hidden behind compute) --
    if (t + 1 < T_TOT) stage_x(t + 1, ldsX[(t + 1) & 1]);

    // ---- deferred coalesced write-back of h_{t-1} -----------------------
    if (t > 0) {
#pragma unroll
      for (int i2 = 0; i2 < 2; ++i2) {
        int i = tid + i2 * 128;
        int r = i >> 4, c = i & 15;
        *(uint2*)(seq + ((b0 + r) * (long)T_TOT + (t - 1)) * HID + c * 4) =
            *(const uint2*)&ldsH[r * LT_STRIDE + c * 4];
      }
    }

    v16bf ax0 = load_a(xbuf, nl, half, 0);
    v16bf ax1 = load_a(xbuf, nl, half, 1);
    v16bf ah0 = load_a(ldsH, nl, half, 0);
    v16bf ah1 = load_a(ldsH, nl, half, 1);
    __syncthreads();                 // C: all LDS reads done before h rewrite

    v8f acc[4];
#pragma unroll
    for (int nt = 0; nt < 4; ++nt) {
      v8f a;
      a = __builtin_amdgcn_wmma_f32_16x16x32_bf16(false, ax0, false, wif[nt][0].b, (short)0, bias_acc[nt], false, false);
      a = __builtin_amdgcn_wmma_f32_16x16x32_bf16(false, ax1, false, wif[nt][1].b, (short)0, a, false, false);
      a = __builtin_amdgcn_wmma_f32_16x16x32_bf16(false, ah0, false, whf[nt][0].b, (short)0, a, false, false);
      a = __builtin_amdgcn_wmma_f32_16x16x32_bf16(false, ah1, false, whf[nt][1].b, (short)0, a, false, false);
      acc[nt] = a;
    }

    // ---- elementwise LSTM update (wave-local columns) -------------------
#pragma unroll
    for (int r = 0; r < 8; ++r) {
      float iv = sigm(acc[0][r]);
      float fv = sigm(acc[1][r]);
      float gv = tanh_f(acc[2][r]);
      float ov = sigm(acc[3][r]);
      float cv = fv * cst[r] + iv * gv;
      cst[r] = cv;
      float hv = ov * tanh_f(cv);
      int m   = r + 8 * half;        // C/D layout: m = r + 8*(lane/16), n = lane%16
      ldsH[m * LT_STRIDE + wv * 16 + nl] = f2bf(hv);
    }
  }

  // ---- final write-back of h_{T-1} --------------------------------------
  __syncthreads();
#pragma unroll
  for (int i2 = 0; i2 < 2; ++i2) {
    int i = tid + i2 * 128;
    int r = i >> 4, c = i & 15;
    *(uint2*)(seq + ((b0 + r) * (long)T_TOT + (T_TOT - 1)) * HID + c * 4) =
        *(const uint2*)&ldsH[r * LT_STRIDE + c * 4];
  }
}

// ---------------------------------------------------------------------------
// Launch
// ---------------------------------------------------------------------------
extern "C" void kernel_launch(void* const* d_in, const int* in_sizes, int n_in,
                              void* d_out, int out_size, void* d_ws, size_t ws_size,
                              hipStream_t stream) {
  (void)in_sizes; (void)n_in; (void)out_size; (void)ws_size;

  const size_t SEQ = (size_t)B_TOT * T_TOT * HID;   // 67,108,864 bf16 = 128 MB
  u16*   seq   = (u16*)d_ws;
  u16*   wip   = seq + SEQ;                         // 4 * 256*64 bf16
  u16*   whp   = wip + 4 * GATES * HID;
  float* biasp = (float*)(whp + 4 * GATES * HID);   // 4 * 256 fp32

  // 1) pad/convert input to bf16 [B][T][64]
  cvt_input_k<<<(int)(SEQ / 256), 256, 0, stream>>>((const float*)d_in[0], seq);

  // 2) pack weights per layer (bf16, K padded to 64) + fold biases
  for (int l = 0; l < 4; ++l) {
    int din = (l == 0) ? 18 : HID;
    pack_k<<<64, 256, 0, stream>>>(
        (const float*)d_in[1 + 4 * l], (const float*)d_in[2 + 4 * l],
        (const float*)d_in[3 + 4 * l], (const float*)d_in[4 + 4 * l],
        wip + (size_t)l * GATES * HID, whp + (size_t)l * GATES * HID,
        biasp + (size_t)l * GATES, din);
  }

  // 3) four LSTM layers, in place on seq
  for (int l = 0; l < 4; ++l) {
    lstm_layer_k<<<B_TOT / 16, 128, 0, stream>>>(
        seq, wip + (size_t)l * GATES * HID, whp + (size_t)l * GATES * HID,
        biasp + (size_t)l * GATES);
  }

  // 4) MLP head on last timestep
  head_k<<<B_TOT, 128, 0, stream>>>(seq,
      (const float*)d_in[17], (const float*)d_in[18],
      (const float*)d_in[19], (const float*)d_in[20],
      (float*)d_out);
}
